// InvariantPointAttentionMultimer_13838384628002
// MI455X (gfx1250) — matrix operand order repaired and checked by reading, not compiled, not run on hardware
//
#include <hip/hip_runtime.h>
#include <hip/hip_bf16.h>

typedef __attribute__((ext_vector_type(16))) __bf16 v16bf;
typedef __attribute__((ext_vector_type(8)))  float  v8f;

#define NRES 768
#define CSD  384
#define CZD  128
#define HD   12
#define CD   16
#define CATD 2112   // H*(CZ+C+4*PV)
#define WMMA_BF16(Cacc, Afrag, Bfrag) \
  __builtin_amdgcn_wmma_f32_16x16x32_bf16(false, (Afrag), false, (Bfrag), (short)0, (Cacc), false, false)

// ---- CDNA5 WMMA fragment layout helpers (wave32) ----
// A 16x32 (MxK) 16-bit: lane m=lane&15, half=lane>>4; elem e -> k per ISA table.
__device__ __forceinline__ int wmma_ka(int e, int half) {
  int v = e >> 1;
  return ((v & 4) << 2) | (half << 3) | ((v & 3) << 1) | (e & 1);
}

__device__ __forceinline__ v16bf frag_a_bf(const __bf16* p, size_t ms, size_t ks, int lane, int kmax) {
  int m = lane & 15, half = lane >> 4;
  v16bf a;
#pragma unroll
  for (int e = 0; e < 16; ++e) {
    int k = wmma_ka(e, half);
    a[e] = (k < kmax) ? p[(size_t)m * ms + (size_t)k * ks] : (__bf16)0.f;
  }
  return a;
}

__device__ __forceinline__ v16bf frag_a_f32(const float* p, size_t ms, size_t ks, int lane, int kmax) {
  int m = lane & 15, half = lane >> 4;
  v16bf a;
#pragma unroll
  for (int e = 0; e < 16; ++e) {
    int k = wmma_ka(e, half);
    a[e] = (k < kmax) ? (__bf16)p[(size_t)m * ms + (size_t)k * ks] : (__bf16)0.f;
  }
  return a;
}

// B 32x16 (KxN) 16-bit: lane n=lane&15, half=lane>>4; elem e -> k per ISA table (same as A).
__device__ __forceinline__ v16bf frag_b_bf(const __bf16* p, size_t ks, size_t ns, int lane, int nmax) {
  int n = lane & 15, half = lane >> 4;
  v16bf b;
#pragma unroll
  for (int e = 0; e < 16; ++e) {
    int k = wmma_ka(e, half);
    b[e] = (n < nmax) ? p[(size_t)k * ks + (size_t)n * ns] : (__bf16)0.f;
  }
  return b;
}

__device__ __forceinline__ v16bf frag_b_f32(const float* p, size_t ks, size_t ns, int lane, int nmax) {
  int n = lane & 15, half = lane >> 4;
  v16bf b;
#pragma unroll
  for (int e = 0; e < 16; ++e) {
    int k = wmma_ka(e, half);
    b[e] = (n < nmax) ? (__bf16)p[(size_t)k * ks + (size_t)n * ns] : (__bf16)0.f;
  }
  return b;
}

// ---------------- Kernel 0: weight conversion ----------------
__global__ void k_convert(const float* Wout, const float* Wb, __bf16* Woutbf, __bf16* Wbbf) {
  int tid = blockIdx.x * blockDim.x + threadIdx.x;
  int stride = gridDim.x * blockDim.x;
  for (int i = tid; i < CATD * CSD; i += stride) Woutbf[i] = (__bf16)Wout[i];
  for (int i = tid; i < CZD * 16; i += stride) {
    int r = i >> 4, c = i & 15;
    Wbbf[i] = (c < HD) ? (__bf16)Wb[r * HD + c] : (__bf16)0.f;
  }
}

// ---------------- Kernel 1: projections + packed features ----------------
// Qext/Kext: [N][H][32] bf16 = [qs(16) | sqrt(pw)*point(12) | 0(4)]  (qs scaled by 1/sqrt(C))
// qoff = -0.5*pw*|qp|^2 ; koff = -0.5*pw*|kp|^2 + bb
// Vbf: [N][H][16] bf16 ; Vpbf: [N][H][24] bf16 (global-frame value points, (p,d) flattened)
__global__ void k_prep(const float* s, const float* rot, const float* trans,
                       const float* Wq, const float* Wk, const float* Wv,
                       const float* Wqp, const float* bqp, const float* Wkp, const float* bkp,
                       const float* Wvp, const float* bvp, const float* bb, const float* hw,
                       __bf16* Qext, __bf16* Kext, float* qoff, float* koff,
                       __bf16* Vbf, __bf16* Vpbf) {
  __shared__ float srow[CSD];
  __shared__ float proj[1152];
  int n = blockIdx.x, tid = threadIdx.x;
  for (int i = tid; i < CSD; i += 256) srow[i] = s[(size_t)n * CSD + i];
  __syncthreads();
  for (int o = tid; o < 1152; o += 256) {
    const float* W; int od, col; float acc;
    if      (o < 192) { W = Wq;  od = 192; col = o;       acc = 0.f; }
    else if (o < 384) { W = Wk;  od = 192; col = o - 192; acc = 0.f; }
    else if (o < 576) { W = Wv;  od = 192; col = o - 384; acc = 0.f; }
    else if (o < 720) { W = Wqp; od = 144; col = o - 576; acc = bqp[col]; }
    else if (o < 864) { W = Wkp; od = 144; col = o - 720; acc = bkp[col]; }
    else              { W = Wvp; od = 288; col = o - 864; acc = bvp[col]; }
    for (int r = 0; r < CSD; ++r) acc += srow[r] * W[(size_t)r * od + col];
    proj[o] = acc;
  }
  __syncthreads();
  float R[9], T[3];
  for (int i = 0; i < 9; ++i) R[i] = rot[(size_t)n * 9 + i];
  for (int i = 0; i < 3; ++i) T[i] = trans[(size_t)n * 3 + i];
  for (int t = tid; t < 696; t += 256) {
    if (t < 192) {                     // qs * 1/sqrt(C)
      int h = t >> 4, c = t & 15;
      Qext[((size_t)n * HD + h) * 32 + c] = (__bf16)(proj[t] * 0.25f);
    } else if (t < 384) {              // ks
      int r = t - 192; int h = r >> 4, c = r & 15;
      Kext[((size_t)n * HD + h) * 32 + c] = (__bf16)proj[t];
    } else if (t < 576) {              // v
      int r = t - 384; int h = r >> 4, c = r & 15;
      Vbf[((size_t)n * HD + h) * 16 + c] = (__bf16)proj[t];
    } else if (t < 588) {              // q-points, head t-576
      int h = t - 576;
      float pwv = log1pf(expf(hw[h])) * 0.23570226f;  // softplus * sqrt(2/(9*PQ))
      float sq = sqrtf(pwv), s2 = 0.f;
      for (int p = 0; p < 4; ++p) {
        float pj[3];
        for (int j = 0; j < 3; ++j) pj[j] = proj[576 + h * 12 + j * 4 + p];
        for (int i = 0; i < 3; ++i) {
          float g = R[i * 3] * pj[0] + R[i * 3 + 1] * pj[1] + R[i * 3 + 2] * pj[2] + T[i];
          s2 += g * g;
          Qext[((size_t)n * HD + h) * 32 + 16 + p * 3 + i] = (__bf16)(sq * g);
        }
      }
      for (int d = 28; d < 32; ++d) Qext[((size_t)n * HD + h) * 32 + d] = (__bf16)0.f;
      qoff[n * HD + h] = -0.5f * pwv * s2;
    } else if (t < 600) {              // k-points, head t-588
      int h = t - 588;
      float pwv = log1pf(expf(hw[h])) * 0.23570226f;
      float sq = sqrtf(pwv), s2 = 0.f;
      for (int p = 0; p < 4; ++p) {
        float pj[3];
        for (int j = 0; j < 3; ++j) pj[j] = proj[720 + h * 12 + j * 4 + p];
        for (int i = 0; i < 3; ++i) {
          float g = R[i * 3] * pj[0] + R[i * 3 + 1] * pj[1] + R[i * 3 + 2] * pj[2] + T[i];
          s2 += g * g;
          Kext[((size_t)n * HD + h) * 32 + 16 + p * 3 + i] = (__bf16)(sq * g);
        }
      }
      for (int d = 28; d < 32; ++d) Kext[((size_t)n * HD + h) * 32 + d] = (__bf16)0.f;
      koff[n * HD + h] = -0.5f * pwv * s2 + bb[h];
    } else {                           // value points
      int idx = t - 600; int h = idx >> 3, p = idx & 7;
      float pj[3];
      for (int j = 0; j < 3; ++j) pj[j] = proj[864 + h * 24 + j * 8 + p];
      for (int i = 0; i < 3; ++i) {
        float g = R[i * 3] * pj[0] + R[i * 3 + 1] * pj[1] + R[i * 3 + 2] * pj[2] + T[i];
        Vpbf[((size_t)n * HD + h) * 24 + p * 3 + i] = (__bf16)g;
      }
    }
  }
}

// ---------------- Kernel 2: logits; z tile staged via async-to-LDS (first z pass) ----------------
__global__ void k_logits(const float* z, const __bf16* Wbbf, const __bf16* Qext, const __bf16* Kext,
                         const float* qoff, const float* koff, const float* mask, float* L) {
  __shared__ __align__(16) float zs[16 * 16 * CZD];  // [q-off][k-off][c] 128 KB
  __shared__ float Lt[16 * 16 * HD];
  int q0 = blockIdx.x * 16, k0 = blockIdx.y * 16;
  int tid = threadIdx.x, wave = tid >> 5, lane = tid & 31;
  int nn = lane & 15, hh = lane >> 4;

  // Stage the 16x16x128 f32 z tile into LDS with GLOBAL_LOAD_ASYNC_TO_LDS_B128 (ASYNCcnt).
  // Each instruction moves 32 lanes x 16B = 512B; 8192 chunks / 128 threads = 64 per thread.
  {
    uint32_t zsBase = (uint32_t)(uintptr_t)zs;  // low 32 bits of generic ptr = LDS byte offset
    for (int it = 0; it < 64; ++it) {
      int cid = it * 128 + tid;
      int r = cid >> 9, c16 = cid & 511;        // r = q-offset row, c16 = 16B chunk within row
      const char* g = (const char*)(z + ((size_t)(q0 + r) * NRES + k0) * CZD) + (size_t)c16 * 16;
      uint32_t l = zsBase + (uint32_t)(cid * 16);
      asm volatile("global_load_async_to_lds_b128 %0, %1, off" :: "v"(l), "v"(g) : "memory");
    }
    asm volatile("s_wait_asynccnt 0x0" ::: "memory");
  }
  __syncthreads();

  // pair bias: rows = 16 k's of query row (q0+mt), cols = heads, K = 128 channels of z (from LDS)
  for (int mt = wave; mt < 16; mt += 4) {
    v8f c = {0.f, 0.f, 0.f, 0.f, 0.f, 0.f, 0.f, 0.f};
    const float* zrow = zs + mt * 16 * CZD;
    for (int cs = 0; cs < 4; ++cs) {
      v16bf A = frag_a_f32(zrow + cs * 32, CZD, 1, lane, 32);
      v16bf B = frag_b_bf(Wbbf + (size_t)cs * 32 * 16, 16, 1, lane, 16);
      c = WMMA_BF16(c, A, B);
    }
#pragma unroll
    for (int v = 0; v < 8; ++v) {
      int m = v + (hh << 3);
      if (nn < HD) Lt[(mt * 16 + m) * HD + nn] = c[v];
    }
  }
  __syncthreads();
  // scalar + point logits: one WMMA per head over 32-wide packed features
  for (int h = wave; h < HD; h += 4) {
    v8f c = {0.f, 0.f, 0.f, 0.f, 0.f, 0.f, 0.f, 0.f};
    v16bf A = frag_a_bf(Qext + ((size_t)q0 * HD + h) * 32, (size_t)HD * 32, 1, lane, 32);
    v16bf B = frag_b_bf(Kext + ((size_t)k0 * HD + h) * 32, 1, (size_t)HD * 32, lane, 16);
    c = WMMA_BF16(c, A, B);
#pragma unroll
    for (int v = 0; v < 8; ++v) {
      int m = v + (hh << 3);
      Lt[(m * 16 + nn) * HD + h] += c[v];
    }
  }
  __syncthreads();
  for (int i = tid; i < 16 * 16 * HD; i += 128) {
    int qi = i / (16 * HD), rem = i % (16 * HD), ki = rem / HD, h = rem % HD;
    int q = q0 + qi, k = k0 + ki;
    float val = (Lt[i] + qoff[q * HD + h] + koff[k * HD + h] +
                 1e5f * (mask[q] * mask[k] - 1.f)) * 0.57735026f;
    L[((size_t)q * NRES + k) * HD + h] = val;
  }
}

// ---------------- Kernel 3: softmax over k (wave per head) ----------------
// Writes TWO layouts so every later A-operand has k contiguous per lane (b128-vectorizable):
//   Aq[h][q][k]           : for o / o_pt  (M = q rows)
//   Ak[q][h(pad16)][k]    : for o_pair    (M = head rows, heads 12..15 zeroed)
__global__ void k_softmax(const float* L, __bf16* Aq, __bf16* Ak) {
  int q = blockIdx.x;
  int wave = threadIdx.x >> 5, lane = threadIdx.x & 31;
  if (wave < HD) {
    int h = wave;
    const float* Lq = L + (size_t)q * NRES * HD + h;
    float mx = -3.0e38f;
    for (int k = lane; k < NRES; k += 32) mx = fmaxf(mx, Lq[(size_t)k * HD]);
#pragma unroll
    for (int o = 16; o > 0; o >>= 1) mx = fmaxf(mx, __shfl_xor(mx, o, 32));
    float sum = 0.f;
    for (int k = lane; k < NRES; k += 32) sum += expf(Lq[(size_t)k * HD] - mx);
#pragma unroll
    for (int o = 16; o > 0; o >>= 1) sum += __shfl_xor(sum, o, 32);
    float inv = 1.f / sum;
    for (int k = lane; k < NRES; k += 32) {
      __bf16 a = (__bf16)(expf(Lq[(size_t)k * HD] - mx) * inv);
      Aq[((size_t)h * NRES + q) * NRES + k] = a;
      Ak[((size_t)q * 16 + h) * NRES + k] = a;
    }
  }
  if (wave < 4) {  // zero head padding rows in Ak so M=16 A-operands need no bounds check
    for (int k = lane; k < NRES; k += 32)
      Ak[((size_t)q * 16 + HD + wave) * NRES + k] = (__bf16)0.f;
  }
}

// ---------------- Kernel 4: aggregations (a@v, a@vp, a^T@z per query), second z pass ----------------
__global__ void k_aggregate(const __bf16* Aq, const __bf16* Ak, const __bf16* Vbf, const __bf16* Vpbf,
                            const float* z, const float* rot, const float* trans, __bf16* catbf) {
  __shared__ float optg[16 * HD * 24];
  int q0 = blockIdx.x * 16;
  int tid = threadIdx.x, wave = tid >> 5, lane = tid & 31;
  int nn = lane & 15, hh = lane >> 4;
  const int NUNITS = 12 + 24 + 128;
  for (int u = wave; u < NUNITS; u += 8) {
    v8f c = {0.f, 0.f, 0.f, 0.f, 0.f, 0.f, 0.f, 0.f};
    if (u < 12) {                 // o[q,h,c] : M=q, N=c, K=k
      int h = u;
      for (int ks = 0; ks < 24; ++ks) {
        v16bf A = frag_a_bf(Aq + ((size_t)h * NRES + q0) * NRES + ks * 32, NRES, 1, lane, 32);
        v16bf B = frag_b_bf(Vbf + ((size_t)ks * 32 * HD + h) * 16, (size_t)HD * 16, 1, lane, 16);
        c = WMMA_BF16(c, A, B);
      }
#pragma unroll
      for (int v = 0; v < 8; ++v) {
        int m = v + (hh << 3);
        catbf[(size_t)(q0 + m) * CATD + h * 16 + nn] = (__bf16)c[v];
      }
    } else if (u < 36) {          // o_pt global frame: M=q, N=(p,d) 24 wide
      int h = (u - 12) >> 1, nt = (u - 12) & 1;
      int nmax = nt ? 8 : 16;
      for (int ks = 0; ks < 24; ++ks) {
        v16bf A = frag_a_bf(Aq + ((size_t)h * NRES + q0) * NRES + ks * 32, NRES, 1, lane, 32);
        v16bf B = frag_b_bf(Vpbf + ((size_t)ks * 32 * HD + h) * 24 + nt * 16, (size_t)HD * 24, 1, lane, nmax);
        c = WMMA_BF16(c, A, B);
      }
#pragma unroll
      for (int v = 0; v < 8; ++v) {
        int m = v + (hh << 3), col = nt * 16 + nn;
        if (col < 24) optg[(m * HD + h) * 24 + col] = c[v];
      }
    } else {                      // o_pair[q] = a_q^T @ z_q : M=heads(padded 16), N=c, K=k
      int qi = (u - 36) >> 3, ct = (u - 36) & 7;
      size_t qg = (size_t)(q0 + qi);
      for (int ks = 0; ks < 24; ++ks) {
        v16bf A = frag_a_bf(Ak + (qg * 16) * NRES + ks * 32, NRES, 1, lane, 32);
        v16bf B = frag_b_f32(z + (qg * NRES + ks * 32) * CZD + ct * 16, CZD, 1, lane, 16);
        c = WMMA_BF16(c, A, B);
      }
#pragma unroll
      for (int v = 0; v < 8; ++v) {
        int h = v + (hh << 3);
        if (h < HD) catbf[qg * CATD + 576 + h * CZD + ct * 16 + nn] = (__bf16)c[v];
      }
    }
  }
  __syncthreads();
  // inverse rigid transform of aggregated points + norm (sum of a == 1, so subtract trans directly)
  for (int i = tid; i < 16 * HD * 8; i += 256) {
    int qi = i / 96, r = i % 96, h = r / 8, p = r & 7;
    int q = q0 + qi;
    float g[3], loc[3];
    for (int d = 0; d < 3; ++d) g[d] = optg[(qi * HD + h) * 24 + p * 3 + d] - trans[(size_t)q * 3 + d];
    for (int d = 0; d < 3; ++d)
      loc[d] = rot[(size_t)q * 9 + d] * g[0] + rot[(size_t)q * 9 + 3 + d] * g[1] + rot[(size_t)q * 9 + 6 + d] * g[2];
    float nr = sqrtf(loc[0] * loc[0] + loc[1] * loc[1] + loc[2] * loc[2] + 1e-8f);
    size_t base = (size_t)q * CATD;
    catbf[base + 192 + h * 8 + p] = (__bf16)loc[0];
    catbf[base + 288 + h * 8 + p] = (__bf16)loc[1];
    catbf[base + 384 + h * 8 + p] = (__bf16)loc[2];
    catbf[base + 480 + h * 8 + p] = (__bf16)nr;
  }
}

// ---------------- Kernel 5: output GEMM cat @ Wout + bout ----------------
__global__ void k_out(const __bf16* catbf, const __bf16* Woutbf, const float* bout, float* out) {
  int q0 = blockIdx.x * 16, n0 = blockIdx.y * 16;
  int lane = threadIdx.x & 31;
  v8f c = {0.f, 0.f, 0.f, 0.f, 0.f, 0.f, 0.f, 0.f};
  for (int ks = 0; ks < CATD / 32; ++ks) {
    v16bf A = frag_a_bf(catbf + (size_t)q0 * CATD + ks * 32, CATD, 1, lane, 32);
    v16bf B = frag_b_bf(Woutbf + (size_t)ks * 32 * CSD + n0, CSD, 1, lane, 16);
    c = WMMA_BF16(c, A, B);
  }
  int nn = lane & 15, hh = lane >> 4;
#pragma unroll
  for (int v = 0; v < 8; ++v) {
    int m = v + (hh << 3);
    out[(size_t)(q0 + m) * CSD + n0 + nn] = c[v] + bout[n0 + nn];
  }
}

extern "C" void kernel_launch(void* const* d_in, const int* in_sizes, int n_in,
                              void* d_out, int out_size, void* d_ws, size_t ws_size,
                              hipStream_t stream) {
  const float* s     = (const float*)d_in[0];
  const float* z     = (const float*)d_in[1];
  const float* rot   = (const float*)d_in[2];
  const float* trans = (const float*)d_in[3];
  const float* mask  = (const float*)d_in[4];
  const float* Wq    = (const float*)d_in[5];
  const float* Wk    = (const float*)d_in[6];
  const float* Wv    = (const float*)d_in[7];
  const float* Wqp   = (const float*)d_in[8];
  const float* bqp   = (const float*)d_in[9];
  const float* Wkp   = (const float*)d_in[10];
  const float* bkp   = (const float*)d_in[11];
  const float* Wvp   = (const float*)d_in[12];
  const float* bvp   = (const float*)d_in[13];
  const float* Wb    = (const float*)d_in[14];
  const float* bb    = (const float*)d_in[15];
  const float* hw    = (const float*)d_in[16];
  const float* Wout  = (const float*)d_in[17];
  const float* bout  = (const float*)d_in[18];
  float* out = (float*)d_out;

  uint8_t* w = (uint8_t*)d_ws;
  size_t off = 0;
  auto take = [&](size_t n) { void* p = w + off; off = (off + n + 255) & ~(size_t)255; return p; };
  __bf16* Woutbf = (__bf16*)take((size_t)CATD * CSD * 2);
  __bf16* Wbbf   = (__bf16*)take((size_t)CZD * 16 * 2);
  __bf16* Qext   = (__bf16*)take((size_t)NRES * HD * 32 * 2);
  __bf16* Kext   = (__bf16*)take((size_t)NRES * HD * 32 * 2);
  float*  qoff   = (float*) take((size_t)NRES * HD * 4);
  float*  koff   = (float*) take((size_t)NRES * HD * 4);
  __bf16* Vbf    = (__bf16*)take((size_t)NRES * HD * 16 * 2);
  __bf16* Vpbf   = (__bf16*)take((size_t)NRES * HD * 24 * 2);
  __bf16* Aq     = (__bf16*)take((size_t)HD * NRES * NRES * 2);
  __bf16* Ak     = (__bf16*)take((size_t)NRES * 16 * NRES * 2);
  float*  L      = (float*) take((size_t)NRES * NRES * HD * 4);
  __bf16* catbf  = (__bf16*)L;  // alias: L is dead after softmax, cat fits inside it

  k_convert<<<256, 256, 0, stream>>>(Wout, Wb, Woutbf, Wbbf);
  k_prep<<<NRES, 256, 0, stream>>>(s, rot, trans, Wq, Wk, Wv, Wqp, bqp, Wkp, bkp,
                                   Wvp, bvp, bb, hw, Qext, Kext, qoff, koff, Vbf, Vpbf);
  k_logits<<<dim3(NRES / 16, NRES / 16), 128, 0, stream>>>(z, Wbbf, Qext, Kext, qoff, koff, mask, L);
  k_softmax<<<NRES, 384, 0, stream>>>(L, Aq, Ak);
  k_aggregate<<<NRES / 16, 256, 0, stream>>>(Aq, Ak, Vbf, Vpbf, z, rot, trans, catbf);
  k_out<<<dim3(NRES / 16, CSD / 16), 32, 0, stream>>>(catbf, Woutbf, bout, out);
}
